// GRUModel_18519898981046
// MI455X (gfx1250) — matrix operand order repaired and checked by reading, not compiled, not run on hardware
//
#include <hip/hip_runtime.h>

// ---------------------------------------------------------------------------
// Fused 2-layer GRU encoder (T=512) + 10-step autoregressive decoder.
// One wave32 owns 16 batch rows; hidden state h (16x64, f16) lives in a
// per-wave LDS tile; recurrent GEMMs run on v_wmma_f32_16x16x32_f16 with
// f32 accumulators; all weights are LDS-resident (pre-swizzled into the
// wave32 WMMA B-fragment layout at kernel start). Nothing recurrent ever
// touches HBM: only x is streamed in and y written out.
// Gate activations are branch-free v_exp_f32/v_rcp_f32 sequences so the
// elementwise tail between WMMAs stays short and EXEC stays all-ones.
// ---------------------------------------------------------------------------

#define HID        64
#define TSTEPS     512
#define DECSTEPS   10
#define WAVES_WG   4
#define THREADS    (WAVES_WG * 32)
#define ROWS_WAVE  16
#define ROWS_WG    (WAVES_WG * ROWS_WAVE)   // 64
#define NBATCH     16384
#define NBLOCKS    (NBATCH / ROWS_WG)       // 256

typedef __attribute__((ext_vector_type(16))) _Float16 v16h;
typedef __attribute__((ext_vector_type(8)))  float    v8f;

// ---- dynamic-LDS layout (byte offsets, all 16B aligned) -------------------
#define OFF_WFRAG  0                          // 3 mats * 24 frags * 512 f16 = 73728 B
#define OFF_WIH0   73728                      // 192*3 f32
#define OFF_BIH0   (OFF_WIH0 + 576 * 4)
#define OFF_BHH0   (OFF_BIH0 + 192 * 4)
#define OFF_BIH1   (OFF_BHH0 + 192 * 4)
#define OFF_BHH1   (OFF_BIH1 + 192 * 4)
#define OFF_WFC    (OFF_BHH1 + 192 * 4)
#define OFF_BFC    (OFF_WFC  + 192 * 4)
#define OFF_HB     (OFF_BFC + 16)             // per wave: h0[16x64] + h1[16x64] f16
#define OFF_XB     (OFF_HB + WAVES_WG * 2048 * 2)
#define SMEM_BYTES (OFF_XB + WAVES_WG * 64 * 4)   // = 97296 B

// Load one 32-byte WMMA fragment (two 16B chunks) from LDS.
// memcpy keeps byte-level aliasing (ordered vs. the f16 gate stores).
__device__ __forceinline__ v16h ld_frag16(const _Float16* p0, const _Float16* p1) {
  v16h r;
  __builtin_memcpy(&r, __builtin_assume_aligned(p0, 16), 16);
  __builtin_memcpy(reinterpret_cast<char*>(&r) + 16, __builtin_assume_aligned(p1, 16), 16);
  return r;
}

// B fragment (32x16, K-major per lane half): lane holds col n = lane%16,
// K = 0..31 split across lane halves. Pre-swizzled at init.
__device__ __forceinline__ v16h ldB(const _Float16* wfrag, int mat, int nT, int kT, int lane) {
  const _Float16* p = wfrag + (((mat * 24 + nT * 2 + kT) * 32 + lane) << 4);
  return ld_frag16(p, p + 8);
}

// A fragment (16x32 f16) from a row-major [16][64] f16 LDS tile, following the
// documented 16-bit A layout: lane<16 -> row=lane, K {0..7,16..23};
// lane>=16 -> row=lane-16, K {8..15,24..31}.
__device__ __forceinline__ v16h ldA(const _Float16* hbuf, int kT, int lane) {
  const _Float16* p0 = hbuf + (lane & 15) * HID + kT * 32 + ((lane >> 4) << 3);
  return ld_frag16(p0, p0 + 16);
}

__device__ __forceinline__ v8f wmma_f16(v16h a, v16h b, v8f c) {
  // (neg_a, A, neg_b, B, c_mod, C, reuse_a, reuse_b)
  return __builtin_amdgcn_wmma_f32_16x16x32_f16(false, a, false, b, (short)0, c, false, false);
}

// Branch-free activations: v_exp_f32 + v_rcp_f32 only (no IEEE div expansion,
// no divergent OCML tanh path). Accuracy ~1ulp rcp + exact exp2: plenty for
// an f16-state GRU.
#define LOG2E 1.4426950408889634f
__device__ __forceinline__ float fast_sigmoid(float x) {
  return __builtin_amdgcn_rcpf(1.0f + __builtin_amdgcn_exp2f(-LOG2E * x));
}
__device__ __forceinline__ float fast_tanh(float x) {
  // tanh(x) = 2*sigmoid(2x) - 1
  return 2.0f * __builtin_amdgcn_rcpf(1.0f + __builtin_amdgcn_exp2f(-2.0f * LOG2E * x)) - 1.0f;
}

// One fused time step for both GRU layers (16 batch rows per wave).
// h?C: persistent f32 hidden state in WMMA C layout (lane = col, vgpr = row).
__device__ __forceinline__ void gru_step(
    const _Float16* wfrag,
    const float* wih0, const float* bih0, const float* bhh0,
    const float* bih1, const float* bhh1,
    _Float16* hbuf0, _Float16* hbuf1, const float* xrow,
    float h0C[4][8], float h1C[4][8], int lane)
{
  const int nlo = lane & 15;
  const int hi  = lane >> 4;

  float xm0[8], xm1[8], xm2[8];
#pragma unroll
  for (int v = 0; v < 8; ++v) {
    const int m = v + hi * 8;
    xm0[v] = xrow[m * 4 + 0];
    xm1[v] = xrow[m * 4 + 1];
    xm2[v] = xrow[m * 4 + 2];
  }

  // old hidden states as A fragments
  v16h a00 = ldA(hbuf0, 0, lane), a01 = ldA(hbuf0, 1, lane);
  v16h a10 = ldA(hbuf1, 0, lane), a11 = ldA(hbuf1, 1, lane);

  // ------------------------- layer 0 -------------------------------------
#pragma unroll
  for (int nt = 0; nt < 4; ++nt) {
    const int cr = nt * 16 + nlo, cz = cr + 64, cn = cr + 128;
    const float br  = bih0[cr] + bhh0[cr];
    const float bz  = bih0[cz] + bhh0[cz];
    const float bnI = bih0[cn];
    const float bnH = bhh0[cn];
    const float wr0 = wih0[cr*3], wr1 = wih0[cr*3+1], wr2 = wih0[cr*3+2];
    const float wz0 = wih0[cz*3], wz1 = wih0[cz*3+1], wz2 = wih0[cz*3+2];
    const float wn0 = wih0[cn*3], wn1 = wih0[cn*3+1], wn2 = wih0[cn*3+2];

    v8f aR, aZ, aN;
    float gin[8];
#pragma unroll
    for (int v = 0; v < 8; ++v) {   // K=3 input GEMM on VALU, folded into C
      aR[v]  = br  + xm0[v]*wr0 + xm1[v]*wr1 + xm2[v]*wr2;
      aZ[v]  = bz  + xm0[v]*wz0 + xm1[v]*wz1 + xm2[v]*wz2;
      gin[v] = bnI + xm0[v]*wn0 + xm1[v]*wn1 + xm2[v]*wn2;
      aN[v]  = bnH;
    }
    aR = wmma_f16(a00, ldB(wfrag, 0, nt,     0, lane), aR);
    aR = wmma_f16(a01, ldB(wfrag, 0, nt,     1, lane), aR);
    aZ = wmma_f16(a00, ldB(wfrag, 0, nt + 4, 0, lane), aZ);
    aZ = wmma_f16(a01, ldB(wfrag, 0, nt + 4, 1, lane), aZ);
    aN = wmma_f16(a00, ldB(wfrag, 0, nt + 8, 0, lane), aN);
    aN = wmma_f16(a01, ldB(wfrag, 0, nt + 8, 1, lane), aN);
#pragma unroll
    for (int v = 0; v < 8; ++v) {
      const float r  = fast_sigmoid(aR[v]);
      const float z  = fast_sigmoid(aZ[v]);
      const float n  = fast_tanh(gin[v] + r * aN[v]);
      const float hn = (1.0f - z) * n + z * h0C[nt][v];
      h0C[nt][v] = hn;
      hbuf0[(v + hi * 8) * HID + cr] = (_Float16)hn;   // C->LDS for A-transpose
    }
  }

  // new h0 re-read as A fragments (layer-1 input); DS is in-order per wave
  v16h b00 = ldA(hbuf0, 0, lane), b01 = ldA(hbuf0, 1, lane);

  // ------------------------- layer 1 -------------------------------------
#pragma unroll
  for (int nt = 0; nt < 4; ++nt) {
    const int cr = nt * 16 + nlo, cz = cr + 64, cn = cr + 128;
    const float br = bih1[cr] + bhh1[cr];
    const float bz = bih1[cz] + bhh1[cz];
    const float bi = bih1[cn];
    const float bh = bhh1[cn];
    v8f aR, aZ, aNi, aNh;
#pragma unroll
    for (int v = 0; v < 8; ++v) { aR[v] = br; aZ[v] = bz; aNi[v] = bi; aNh[v] = bh; }

    aR  = wmma_f16(b00, ldB(wfrag, 1, nt,     0, lane), aR);
    aR  = wmma_f16(b01, ldB(wfrag, 1, nt,     1, lane), aR);
    aR  = wmma_f16(a10, ldB(wfrag, 2, nt,     0, lane), aR);
    aR  = wmma_f16(a11, ldB(wfrag, 2, nt,     1, lane), aR);
    aZ  = wmma_f16(b00, ldB(wfrag, 1, nt + 4, 0, lane), aZ);
    aZ  = wmma_f16(b01, ldB(wfrag, 1, nt + 4, 1, lane), aZ);
    aZ  = wmma_f16(a10, ldB(wfrag, 2, nt + 4, 0, lane), aZ);
    aZ  = wmma_f16(a11, ldB(wfrag, 2, nt + 4, 1, lane), aZ);
    aNi = wmma_f16(b00, ldB(wfrag, 1, nt + 8, 0, lane), aNi);
    aNi = wmma_f16(b01, ldB(wfrag, 1, nt + 8, 1, lane), aNi);
    aNh = wmma_f16(a10, ldB(wfrag, 2, nt + 8, 0, lane), aNh);
    aNh = wmma_f16(a11, ldB(wfrag, 2, nt + 8, 1, lane), aNh);
#pragma unroll
    for (int v = 0; v < 8; ++v) {
      const float r  = fast_sigmoid(aR[v]);
      const float z  = fast_sigmoid(aZ[v]);
      const float n  = fast_tanh(aNi[v] + r * aNh[v]);
      const float hn = (1.0f - z) * n + z * h1C[nt][v];
      h1C[nt][v] = hn;
      hbuf1[(v + hi * 8) * HID + cr] = (_Float16)hn;
    }
  }
}

__global__ __launch_bounds__(THREADS) void gru_seq2seq_kernel(
    const float* __restrict__ x,
    const float* __restrict__ W_ih0, const float* __restrict__ W_hh0,
    const float* __restrict__ b_ih0, const float* __restrict__ b_hh0,
    const float* __restrict__ W_ih1, const float* __restrict__ W_hh1,
    const float* __restrict__ b_ih1, const float* __restrict__ b_hh1,
    const float* __restrict__ W_fc,  const float* __restrict__ b_fc,
    float* __restrict__ out)
{
  extern __shared__ char smem[];
  _Float16* wfrag = (_Float16*)(smem + OFF_WFRAG);
  float* s_wih0 = (float*)(smem + OFF_WIH0);
  float* s_bih0 = (float*)(smem + OFF_BIH0);
  float* s_bhh0 = (float*)(smem + OFF_BHH0);
  float* s_bih1 = (float*)(smem + OFF_BIH1);
  float* s_bhh1 = (float*)(smem + OFF_BHH1);
  float* s_wfc  = (float*)(smem + OFF_WFC);
  float* s_bfc  = (float*)(smem + OFF_BFC);
  _Float16* hb  = (_Float16*)(smem + OFF_HB);
  float* s_xb   = (float*)(smem + OFF_XB);

  const int tid  = threadIdx.x;
  const int lane = tid & 31;
  const int wave = tid >> 5;

  // ---- init: swizzle weights into WMMA B-fragment layout (f32 -> f16) ----
  // GEMM is h @ W^T, so B[k][n] = W[n*64 + k].
  for (int idx = tid; idx < 3 * 24 * 32 * 16; idx += THREADS) {
    const int e   = idx & 15;
    const int l   = (idx >> 4) & 31;
    const int f   = idx >> 9;
    const int mat = f / 24;
    const int rem = f % 24;
    const int nT  = rem >> 1;
    const int kT  = rem & 1;
    const float* W = (mat == 0) ? W_hh0 : ((mat == 1) ? W_ih1 : W_hh1);
    const int n = nT * 16 + (l & 15);
    const int k = kT * 32 + (l >> 4) * 8 + (e & 7) + (e >> 3) * 16;
    wfrag[idx] = (_Float16)W[n * HID + k];
  }
  for (int i = tid; i < 576; i += THREADS) s_wih0[i] = W_ih0[i];
  for (int i = tid; i < 192; i += THREADS) {
    s_bih0[i] = b_ih0[i]; s_bhh0[i] = b_hh0[i];
    s_bih1[i] = b_ih1[i]; s_bhh1[i] = b_hh1[i];
    s_wfc[i]  = W_fc[i];
  }
  if (tid < 3) s_bfc[tid] = b_fc[tid];
  for (int i = tid; i < WAVES_WG * 2048; i += THREADS) hb[i] = (_Float16)0.0f;
  __syncthreads();

  _Float16* hbuf0 = hb + wave * 2048;
  _Float16* hbuf1 = hbuf0 + 1024;
  float* xrow = s_xb + wave * 64;

  float h0C[4][8], h1C[4][8];
#pragma unroll
  for (int a = 0; a < 4; ++a)
#pragma unroll
    for (int b = 0; b < 8; ++b) { h0C[a][b] = 0.0f; h1C[a][b] = 0.0f; }

  const long bbase = (long)blockIdx.x * ROWS_WG + (long)wave * ROWS_WAVE;

  // ----------------------------- encoder ----------------------------------
  for (int t = 0; t < TSTEPS; ++t) {
    if (lane < 16) {
      const float* xp = x + ((bbase + lane) * (long)TSTEPS + t) * 3;
      xrow[lane * 4 + 0] = xp[0];
      xrow[lane * 4 + 1] = xp[1];
      xrow[lane * 4 + 2] = xp[2];
      __builtin_prefetch(xp + 3, 0, 1);   // next time step of this row
    }
    __builtin_amdgcn_wave_barrier();
    gru_step(wfrag, s_wih0, s_bih0, s_bhh0, s_bih1, s_bhh1,
             hbuf0, hbuf1, xrow, h0C, h1C, lane);
  }

  // ------------------------ autoregressive decoder ------------------------
  for (int s = 0; s < DECSTEPS; ++s) {
    if (lane < 16) {
      // y = fc(h1) BEFORE stepping (matches reference), fed back as input
      const _Float16* hr = hbuf1 + lane * HID;
      float y0 = s_bfc[0], y1 = s_bfc[1], y2 = s_bfc[2];
#pragma unroll 8
      for (int j = 0; j < HID; ++j) {
        const float hv = (float)hr[j];
        y0 += hv * s_wfc[j];
        y1 += hv * s_wfc[64 + j];
        y2 += hv * s_wfc[128 + j];
      }
      float* op = out + ((bbase + lane) * DECSTEPS + s) * 3;
      op[0] = y0; op[1] = y1; op[2] = y2;
      xrow[lane * 4 + 0] = y0;
      xrow[lane * 4 + 1] = y1;
      xrow[lane * 4 + 2] = y2;
    }
    __builtin_amdgcn_wave_barrier();
    gru_step(wfrag, s_wih0, s_bih0, s_bhh0, s_bih1, s_bhh1,
             hbuf0, hbuf1, xrow, h0C, h1C, lane);
  }
}

extern "C" void kernel_launch(void* const* d_in, const int* in_sizes, int n_in,
                              void* d_out, int out_size, void* d_ws, size_t ws_size,
                              hipStream_t stream) {
  const float* x     = (const float*)d_in[0];
  const float* W_ih0 = (const float*)d_in[1];
  const float* W_hh0 = (const float*)d_in[2];
  const float* b_ih0 = (const float*)d_in[3];
  const float* b_hh0 = (const float*)d_in[4];
  const float* W_ih1 = (const float*)d_in[5];
  const float* W_hh1 = (const float*)d_in[6];
  const float* b_ih1 = (const float*)d_in[7];
  const float* b_hh1 = (const float*)d_in[8];
  const float* W_fc  = (const float*)d_in[9];
  const float* b_fc  = (const float*)d_in[10];
  float* out = (float*)d_out;

  // ~95 KB dynamic LDS per WG (WGP budget is 320 KB): raise the cap.
  // Idempotent + stream-independent, so graph-capture safe; result ignored.
  (void)hipFuncSetAttribute((const void*)gru_seq2seq_kernel,
                            hipFuncAttributeMaxDynamicSharedMemorySize,
                            (int)SMEM_BYTES);

  gru_seq2seq_kernel<<<dim3(NBLOCKS), dim3(THREADS), SMEM_BYTES, stream>>>(
      x, W_ih0, W_hh0, b_ih0, b_hh0, W_ih1, W_hh1, b_ih1, b_hh1, W_fc, b_fc, out);
}
